// FFTLayer_47966194762375
// MI455X (gfx1250) — matrix operand order repaired and checked by reading, not compiled, not run on hardware
//
#include <hip/hip_runtime.h>

// MI455X / gfx1250: wave32, WMMA 16x16x32 f16 -> f32, TDM tile staging
typedef __attribute__((ext_vector_type(16))) _Float16 v16h;
typedef __attribute__((ext_vector_type(8)))  float    v8f;
typedef __attribute__((ext_vector_type(4)))  float    f4;
typedef __attribute__((ext_vector_type(4)))  unsigned int v4u;
typedef __attribute__((ext_vector_type(8)))  int      v8i;
typedef __attribute__((ext_vector_type(4)))  int      v4i;

#define FFT_N   512
#define BATCH   256
#define FRAMES  256
#define M_TOTAL (BATCH * FRAMES)   // 65536 rows
#define BM      32                 // M rows per block (A read from HBM exactly once)
#define KT      32                 // WMMA K per step
#define KBLK    (FFT_N / KT)       // 16 K-steps
// TDM pad: 4 DWORDs inserted every 256 DWORDs -> row stride 512+8 = 520 floats,
// 16B alignment preserved for ds_load_b128 fragment reads.
#define LDA_P   520

// ---------------------------------------------------------------------------
// B'[k_in, k_out] = w[k_in]*cos(2*pi*k_in*k_out/512), f16, stored directly in
// the per-lane WMMA B-fragment order used here (32x16 B, contiguous k):
//   fragment id = n_tile*16 + k_block ; per fragment: 32 lanes x 16 halves
//   lane l: column n = n_tile*16 + (l&15); half h -> K = k_block*32 + (l>>4)*16 + h
// ---------------------------------------------------------------------------
__global__ void build_dft_b(const float* __restrict__ win, _Float16* __restrict__ Bf) {
    int id   = blockIdx.x * blockDim.x + threadIdx.x;  // 0 .. 512*512-1
    int h    = id & 15;
    int lane = (id >> 4) & 31;
    int frag = id >> 9;            // 0..511
    int nt   = frag >> 4;          // n-tile 0..31
    int kb   = frag & 15;          // k-block 0..15
    int n    = nt * 16 + (lane & 15);            // output bin
    int k    = kb * 32 + ((lane >> 4) << 4) + h; // input sample (windowed dim)
    int t    = (n * k) & (FFT_N - 1);            // angle periodicity
    float v  = win[k] * cosf(6.28318530717958647692f * (float)t / (float)FFT_N);
    Bf[id]   = (_Float16)v;
}

// ---------------------------------------------------------------------------
// D[32 x 512] = A[32 x 512] (f32) x B' (f16 fragments), per block. 8 wave32s;
// wave w owns columns [w*64, w*64+64). A tile staged to LDS by the TDM.
// ---------------------------------------------------------------------------
__global__ void __launch_bounds__(256) fft_wmma_kernel(
        const float* __restrict__ A,
        const _Float16* __restrict__ Bf,
        float* __restrict__ out) {
    extern __shared__ float As[];                 // [BM][LDA_P] f32 tile (HW-padded)
    const int tid   = threadIdx.x;
    const int mbase = blockIdx.x * BM;
    const int lane  = tid & 31;     // wave32
    const int wave  = tid >> 5;     // 0..7

#if __has_builtin(__builtin_amdgcn_tensor_load_to_lds) && \
    __has_builtin(__builtin_amdgcn_s_wait_tensorcnt)
    // ---- Tensor Data Mover: DMA the 32x512 f32 tile into LDS -----------------
    if (wave == 0) {                               // wave-uniform: one issue per block
        unsigned long long ga =
            (unsigned long long)(uintptr_t)(A + (size_t)mbase * FFT_N);
        unsigned int ldsb = (unsigned int)(uintptr_t)(void*)As;  // LDS byte offset

        // D# group0: count=1 (valid), lds_addr, 57-bit global_addr, type=2
        v4u g0;
        g0[0] = 1u;
        g0[1] = ldsb;
        g0[2] = (unsigned int)ga;
        g0[3] = (unsigned int)((ga >> 32) & 0x01FFFFFFu) | (2u << 30);

        // D# group1: data_size=4B(2), pad_enable, pad_interval=7 (256 DW),
        // pad_amount code 3 (4 DW); tensor 512x32, tile 512x32, dim0_stride=512
        v8i g1;
        g1[0] = (int)((2u << 16) | (1u << 20) | (7u << 22) | (3u << 25));
        g1[1] = (int)(512u << 16);        // tensor_dim0[15:0] -> bits[31:16]
        g1[2] = (int)(32u  << 16);        // tensor_dim0 hi=0 | tensor_dim1[15:0]=32
        g1[3] = (int)(512u << 16);        // tensor_dim1 hi=0 | tile_dim0=512
        g1[4] = (int)32u;                 // tile_dim1=32 | tile_dim2=0
        g1[5] = (int)512u;                // tensor_dim0_stride lo
        g1[6] = 0;
        g1[7] = 0;

        v4i gz = (v4i){0, 0, 0, 0};       // groups 2/3 unused (<=2D tile)
#if __clang_major__ >= 23
        v8i gz8 = (v8i){0, 0, 0, 0, 0, 0, 0, 0};
        __builtin_amdgcn_tensor_load_to_lds(g0, g1, gz, gz, gz8, 0);
#else
        __builtin_amdgcn_tensor_load_to_lds(g0, g1, gz, gz, 0);
#endif
        __builtin_amdgcn_s_wait_tensorcnt(0);      // TENSORcnt==0 before sharing
    }
    __syncthreads();
#else
    // ---- fallback: cooperative b128 staging with identical padded layout -----
    {
        const f4* Ag = (const f4*)(A + (size_t)mbase * FFT_N);
        #pragma unroll
        for (int it = 0; it < 16; ++it) {
            int p   = tid + it * 256;      // 0..4095 float4 slots
            int row = p >> 7;              // 128 float4 per row
            int cf4 = p & 127;
            f4 v    = Ag[row * 128 + cf4];
            *(f4*)(&As[row * LDA_P + cf4 * 4 + 4 * (cf4 >> 6)]) = v;
        }
    }
    __syncthreads();
#endif

    const int lh  = lane & 15;
    const int sel = lane >> 4;      // lane half: A K-phase / C row half

    v8f acc[2][4];
    #pragma unroll
    for (int i = 0; i < 2; ++i)
        #pragma unroll
        for (int j = 0; j < 4; ++j)
            acc[i][j] = (v8f){0.f,0.f,0.f,0.f,0.f,0.f,0.f,0.f};

    for (int kb = 0; kb < KBLK; ++kb) {
        // prefetch next K-step's first B fragment (global_prefetch_b8 path)
        if (kb + 1 < KBLK) {
            const _Float16* nb = Bf + ((size_t)((wave * 4) * 16 + kb + 1) * 32 + lane) * 16;
            __builtin_prefetch(nb, 0, 0);
        }

        // ---- A fragments (16x32 16-bit A, contiguous k order) -------------------------
        // lane<16: K = 0..15 ; lane>=16: K = 16..31 (contiguous, matches B)
        const int k0 = kb * KT + sel * 16;
        const int k1 = k0 + 8;
        const int p0 = k0 + 4 * (k0 >> 8);   // HW pad: +4 floats past DWORD 256
        const int p1 = k1 + 4 * (k1 >> 8);
        v16h a[2];
        #pragma unroll
        for (int i = 0; i < 2; ++i) {
            const float* r = &As[(i * 16 + lh) * LDA_P];
            f4 f0 = *(const f4*)(r + p0);
            f4 f1 = *(const f4*)(r + p0 + 4);
            f4 f2 = *(const f4*)(r + p1);
            f4 f3 = *(const f4*)(r + p1 + 4);
            v16h av;
            av[0]  = (_Float16)f0.x; av[1]  = (_Float16)f0.y;
            av[2]  = (_Float16)f0.z; av[3]  = (_Float16)f0.w;
            av[4]  = (_Float16)f1.x; av[5]  = (_Float16)f1.y;
            av[6]  = (_Float16)f1.z; av[7]  = (_Float16)f1.w;
            av[8]  = (_Float16)f2.x; av[9]  = (_Float16)f2.y;
            av[10] = (_Float16)f2.z; av[11] = (_Float16)f2.w;
            av[12] = (_Float16)f3.x; av[13] = (_Float16)f3.y;
            av[14] = (_Float16)f3.z; av[15] = (_Float16)f3.w;
            a[i] = av;
        }

        // ---- B fragments (pre-swizzled, one 32B coalesced load per lane) -----
        #pragma unroll
        for (int j = 0; j < 4; ++j) {
            const int frag = (wave * 4 + j) * 16 + kb;
            v16h b = *(const v16h*)(Bf + ((size_t)frag * 32 + lane) * 16);
            acc[0][j] = __builtin_amdgcn_wmma_f32_16x16x32_f16(
                false, a[0], false, b, (short)0, acc[0][j], false, false);
            acc[1][j] = __builtin_amdgcn_wmma_f32_16x16x32_f16(
                false, a[1], false, b, (short)0, acc[1][j], false, false);
        }
    }

    // ---- epilogue: C/D layout (VGPR v -> row v + 8*sel, col = lane%16) -------
    #pragma unroll
    for (int i = 0; i < 2; ++i) {
        #pragma unroll
        for (int j = 0; j < 4; ++j) {
            const int col = wave * 64 + j * 16 + lh;
            #pragma unroll
            for (int v = 0; v < 8; ++v) {
                const int row = mbase + i * 16 + v + sel * 8;
                out[(size_t)row * FFT_N + col] = acc[i][j][v];
            }
        }
    }
}

// ---------------------------------------------------------------------------
extern "C" void kernel_launch(void* const* d_in, const int* in_sizes, int n_in,
                              void* d_out, int out_size, void* d_ws, size_t ws_size,
                              hipStream_t stream) {
    const float* x   = (const float*)d_in[0];   // (256,256,512) f32
    const float* win = (const float*)d_in[1];   // (512,) f32
    float* out       = (float*)d_out;           // (256,256,512,1) f32
    _Float16* Bf     = (_Float16*)d_ws;         // 512*512 f16 = 512 KB scratch

    // 1) windowed-DFT matrix, pre-swizzled into WMMA B fragments (L2-resident)
    build_dft_b<<<(FFT_N * FFT_N) / 256, 256, 0, stream>>>(win, Bf);

    // 2) streaming GEMM: 2048 blocks x (32 rows x 512 cols), A read once
    const size_t shmem = (size_t)BM * LDA_P * sizeof(float);  // ~65 KB (<320 KB/WGP)
    fft_wmma_kernel<<<M_TOTAL / BM, 256, shmem, stream>>>(x, Bf, out);
}
